// Multihead_Attn_21741124452520
// MI455X (gfx1250) — matrix operand order repaired and checked by reading, not compiled, not run on hardware
//
#include <hip/hip_runtime.h>

// ---------------------------------------------------------------------------
// Multihead attention block (GN -> kqv 1x1 -> pool -> attn -> upsample -> o 1x1 -> +X)
// Pooling commutes with the 1x1 convs and the GN affine, so every GEMM runs at
// pooled 16x16 resolution. GEMMs use v_wmma_f32_16x16x32_bf16 (wave32 WMMA).
// ---------------------------------------------------------------------------

#define BB   32
#define CC   512
#define HW   4096      // 64*64
#define POS  256       // 16*16 pooled positions
#define HEADS 8
#define HD   64        // channels per head
#define KQV  1536      // 3*C

typedef unsigned short u16;
typedef __bf16 bf16_t;
typedef __attribute__((ext_vector_type(16))) bf16_t v16bf;
typedef __attribute__((ext_vector_type(8)))  float  v8f;

__device__ __forceinline__ u16 f2bf(float x) {
    union { float f; unsigned int u; } c; c.f = x;
    unsigned int u = c.u;
    u = (u + 0x7FFFu + ((u >> 16) & 1u)) >> 16;
    return (u16)u;
}

__device__ __forceinline__ v8f zero8() {
    v8f z;
#pragma unroll
    for (int i = 0; i < 8; ++i) z[i] = 0.0f;
    return z;
}

// A fragment (16x32 bf16, MxK), source row-major [M][lda], bf16.
// Lane l: M = mBase + (l&15); half = l>>4.
// element i: k = kBase + (i>>3)*16 + half*8 + (i&7)
__device__ __forceinline__ v16bf load_afrag(const u16* __restrict__ A, int lda,
                                            int mBase, int kBase, int lane) {
    int m = mBase + (lane & 15);
    int half = lane >> 4;
    const u16* row = A + (size_t)m * lda + kBase + (half << 3);
    union { v16bf v; u16 u[16]; } f;
#pragma unroll
    for (int i = 0; i < 16; ++i) f.u[i] = row[((i >> 3) << 4) + (i & 7)];
    return f.v;
}

// A fragment built from f32 source (converted on the fly).
__device__ __forceinline__ v16bf load_afrag_f32(const float* __restrict__ A, int lda,
                                                int mBase, int kBase, int lane) {
    int m = mBase + (lane & 15);
    int half = lane >> 4;
    const float* row = A + (size_t)m * lda + kBase + (half << 3);
    union { v16bf v; u16 u[16]; } f;
#pragma unroll
    for (int i = 0; i < 16; ++i) f.u[i] = f2bf(row[((i >> 3) << 4) + (i & 7)]);
    return f.v;
}

// B fragment (32x16 bf16, KxN) from B^T stored row-major [N][ldb].
// Lane l: N = nBase + (l&15); k = kBase + (i>>3)*16 + (l>>4)*8 + (i&7)
__device__ __forceinline__ v16bf load_bfrag(const u16* __restrict__ Bt, int ldb,
                                            int nBase, int kBase, int lane) {
    int n = nBase + (lane & 15);
    const u16* row = Bt + (size_t)n * ldb + kBase + ((lane >> 4) << 3);
    union { v16bf v; u16 u[16]; } f;
#pragma unroll
    for (int i = 0; i < 16; ++i) f.u[i] = row[((i >> 3) << 4) + (i & 7)];
    return f.v;
}

__device__ __forceinline__ v8f wmma_bf16(v16bf a, v16bf b, v8f c) {
    return __builtin_amdgcn_wmma_f32_16x16x32_bf16(false, a, false, b,
                                                   (short)0, c, false, false);
}

// ---------------------------------------------------------------------------
// K1: per (b,c) plane: 4x4 average pool -> pooledX[b][p][c], and channel
// sum / sumsq for GroupNorm stats. This is one of the two HBM-bound passes.
__global__ void pool_stats_kernel(const float* __restrict__ X,
                                  float* __restrict__ pooledX,
                                  float* __restrict__ chsum,
                                  float* __restrict__ chsumsq) {
    int bc = blockIdx.x;                 // b*512 + c
    const float* plane = X + (size_t)bc * HW;
    int p = threadIdx.x;                 // pooled cell 0..255
    // each thread prefetches one 64B line -> covers the 16KB plane
    __builtin_prefetch(plane + p * 16, 0, 0);
    int r1 = p >> 4, r2 = p & 15;
    float s = 0.f, sq = 0.f;
#pragma unroll
    for (int i = 0; i < 4; ++i) {
#pragma unroll
        for (int j = 0; j < 4; ++j) {
            float v = plane[(r1 * 4 + i) * 64 + r2 * 4 + j];
            s += v; sq += v * v;
        }
    }
    int b = bc >> 9, c = bc & 511;
    pooledX[((size_t)b * POS + p) * CC + c] = s * 0.0625f;

    __shared__ float sh[256], sh2[256];
    sh[p] = s; sh2[p] = sq;
    __syncthreads();
    for (int off = 128; off > 0; off >>= 1) {
        if (p < off) { sh[p] += sh[p + off]; sh2[p] += sh2[p + off]; }
        __syncthreads();
    }
    if (p == 0) { chsum[bc] = sh[0]; chsumsq[bc] = sh2[0]; }
}

// K2: per (b,group) stats -> per (b,c) scale/shift folding GN affine.
__global__ void group_stats_kernel(const float* __restrict__ chsum,
                                   const float* __restrict__ chsumsq,
                                   const float* __restrict__ gn_w,
                                   const float* __restrict__ gn_b,
                                   float* __restrict__ sc,
                                   float* __restrict__ sh) {
    int g = blockIdx.x * blockDim.x + threadIdx.x;
    if (g >= BB * 128) return;
    int b = g >> 7, gr = g & 127;
    int c0 = gr * 4;
    float su = 0.f, sq = 0.f;
#pragma unroll
    for (int k = 0; k < 4; ++k) {
        su += chsum[b * CC + c0 + k];
        sq += chsumsq[b * CC + c0 + k];
    }
    const float invN = 1.0f / 16384.0f;
    float mu = su * invN;
    float var = sq * invN - mu * mu;
    float rs = rsqrtf(var + 1e-5f);
#pragma unroll
    for (int k = 0; k < 4; ++k) {
        int c = c0 + k;
        sc[b * CC + c] = gn_w[c] * rs;
        sh[b * CC + c] = gn_b[c] - mu * rs * gn_w[c];
    }
}

// K3: apply per-channel affine, emit pooled-normalized activations bf16 [b][p][c].
__global__ void norm_bf16_kernel(const float* __restrict__ pooledX,
                                 const float* __restrict__ sc,
                                 const float* __restrict__ sh,
                                 u16* __restrict__ xnp) {
    size_t idx = (size_t)blockIdx.x * blockDim.x + threadIdx.x;
    if (idx >= (size_t)BB * POS * CC) return;
    int c = (int)(idx & 511);
    int b = (int)(idx >> 17);            // / (256*512)
    xnp[idx] = f2bf(sc[b * CC + c] * pooledX[idx] + sh[b * CC + c]);
}

// Kw: convert weights to bf16 (row-major kept).
__global__ void cvt_w_kernel(const float* __restrict__ kqv_w,
                             const float* __restrict__ o_w,
                             u16* __restrict__ wk, u16* __restrict__ ow) {
    size_t idx = (size_t)blockIdx.x * blockDim.x + threadIdx.x;
    const size_t NK = (size_t)KQV * CC;
    if (idx < NK) wk[idx] = f2bf(kqv_w[idx]);
    else if (idx < NK + (size_t)CC * CC) {
        size_t j = idx - NK;
        ow[j] = f2bf(o_w[j]);
    }
}

// Generic WMMA GEMM: C[b][m][n] = sum_k A[m][k]*Bt[b][n][k] (+bias[m]).
// One wave per 32x32 macro-tile (2x2 WMMA tiles -> 4 WMMA per k-step from
// 4 fragment loads), 8 waves per block, blockIdx.y = batch.
__global__ void gemm_bf16_kernel(const u16* __restrict__ A,
                                 const u16* __restrict__ Bt,
                                 float* __restrict__ C,
                                 const float* __restrict__ bias,
                                 int M, int N, int K,
                                 size_t btStride, size_t cStride) {
    int lane = threadIdx.x & 31;
    int wave = threadIdx.x >> 5;
    int tilesN = N >> 5;                       // macro tiles along N
    int totalTiles = (M >> 5) * tilesN;
    int tile = blockIdx.x * 8 + wave;
    if (tile >= totalTiles) return;
    int mBase = (tile / tilesN) << 5;
    int nBase = (tile % tilesN) << 5;
    int b = blockIdx.y;
    const u16* bt = Bt + (size_t)b * btStride;
    float* c = C + (size_t)b * cStride;

    v8f acc00 = zero8(), acc01 = zero8(), acc10 = zero8(), acc11 = zero8();
    for (int k = 0; k < K; k += 32) {
        v16bf a0 = load_afrag(A, K, mBase,      k, lane);
        v16bf a1 = load_afrag(A, K, mBase + 16, k, lane);
        v16bf b0 = load_bfrag(bt, K, nBase,      k, lane);
        v16bf b1 = load_bfrag(bt, K, nBase + 16, k, lane);
        acc00 = wmma_bf16(a0, b0, acc00);
        acc01 = wmma_bf16(a0, b1, acc01);
        acc10 = wmma_bf16(a1, b0, acc10);
        acc11 = wmma_bf16(a1, b1, acc11);
    }
    int half = lane >> 4;
    int nlo = lane & 15;
#pragma unroll
    for (int r = 0; r < 8; ++r) {
        int m0 = mBase + r + (half << 3);
        int m1 = m0 + 16;
        float bias0 = bias ? bias[m0] : 0.0f;
        float bias1 = bias ? bias[m1] : 0.0f;
        c[(size_t)m0 * N + nBase + nlo]      = acc00[r] + bias0;
        c[(size_t)m0 * N + nBase + 16 + nlo] = acc01[r] + bias0;
        c[(size_t)m1 * N + nBase + nlo]      = acc10[r] + bias1;
        c[(size_t)m1 * N + nBase + 16 + nlo] = acc11[r] + bias1;
    }
}

// K5: scores = (Q^T K)/sqrt(C), softmax rows, write attn probs bf16 [b,h][d][D].
// Block: 256 threads = 8 waves, each wave owns 16 rows (d). grid (2,8,32).
__global__ void attn_softmax_kernel(const float* __restrict__ kqvp,
                                    u16* __restrict__ attnW) {
    int dblk = blockIdx.x, h = blockIdx.y, b = blockIdx.z;
    __shared__ u16 KT[POS][HD];      // K^T : [D][c]  32KB
    __shared__ u16 QT[128][HD];      // Q^T strip : [d][c] 16KB
    const float* base = kqvp + (size_t)b * KQV * POS;
    const float* Kp = base + (size_t)(h * HD) * POS;
    const float* Qp = base + (size_t)(CC + h * HD) * POS;
    int tid = threadIdx.x;
    int dbase = dblk * 128;

    for (int idx = tid; idx < HD * POS; idx += 256) {
        int c = idx >> 8, p = idx & 255;
        KT[p][c] = f2bf(Kp[(size_t)c * POS + p]);
    }
    for (int idx = tid; idx < HD * 128; idx += 256) {
        int c = idx >> 7, dl = idx & 127;
        QT[dl][c] = f2bf(Qp[(size_t)c * POS + dbase + dl]);
    }
    __syncthreads();

    int lane = tid & 31, wave = tid >> 5;
    int half = lane >> 4;
    v16bf a0 = load_afrag(&QT[0][0], HD, wave * 16, 0, lane);
    v16bf a1 = load_afrag(&QT[0][0], HD, wave * 16, 32, lane);

    v8f acc[16];
#pragma unroll
    for (int t = 0; t < 16; ++t) acc[t] = zero8();
#pragma unroll
    for (int t = 0; t < 16; ++t) {
        v16bf b0 = load_bfrag(&KT[0][0], HD, t * 16, 0, lane);
        v16bf b1 = load_bfrag(&KT[0][0], HD, t * 16, 32, lane);
        acc[t] = wmma_bf16(a0, b0, acc[t]);
        acc[t] = wmma_bf16(a1, b1, acc[t]);
    }

    const float scale = 0.044194173824159216f;  // 1/sqrt(512)
    u16* outBase = attnW + ((size_t)(b * HEADS + h) * POS) * POS;
#pragma unroll
    for (int r = 0; r < 8; ++r) {
        float mx = -3.4e38f;
#pragma unroll
        for (int t = 0; t < 16; ++t) {
            acc[t][r] *= scale;
            mx = fmaxf(mx, acc[t][r]);
        }
        // reduce across the 16 lanes of this wave-half (rows live per-half)
#pragma unroll
        for (int off = 1; off < 16; off <<= 1)
            mx = fmaxf(mx, __shfl_xor(mx, off, 32));
        float sum = 0.f;
#pragma unroll
        for (int t = 0; t < 16; ++t) {
            float e = __expf(acc[t][r] - mx);
            acc[t][r] = e;
            sum += e;
        }
#pragma unroll
        for (int off = 1; off < 16; off <<= 1)
            sum += __shfl_xor(sum, off, 32);
        float inv = 1.0f / sum;
        int d = dbase + wave * 16 + r + (half << 3);
        u16* row = outBase + (size_t)d * POS;
#pragma unroll
        for (int t = 0; t < 16; ++t)
            row[t * 16 + (lane & 15)] = f2bf(acc[t][r] * inv);
    }
}

// K6: out[c][d] = sum_D V[c][D]*attn[d][D]; store bf16 transposed attB[b][d][ch].
// grid (8 heads, 32 batch). Each wave owns one c-tile (mt = wave>>1) and 8
// d-tiles, so the f32->bf16 converted V fragment is loaded once per k-step
// and feeds 8 WMMAs.
__global__ void attn_v_kernel(const float* __restrict__ kqvp,
                              const u16* __restrict__ attnW,
                              u16* __restrict__ attB) {
    int h = blockIdx.x, b = blockIdx.y;
    int lane = threadIdx.x & 31, wave = threadIdx.x >> 5;
    int half = lane >> 4;
    int mt = wave >> 1;                 // c-tile 0..3
    int ntBase = (wave & 1) * 8;        // d-tiles ntBase..ntBase+7
    const float* Vp = kqvp + ((size_t)b * KQV + 2 * CC + h * HD) * POS;
    const u16* aw = attnW + ((size_t)(b * HEADS + h) * POS) * POS;

    v8f acc[8];
#pragma unroll
    for (int j = 0; j < 8; ++j) acc[j] = zero8();

    for (int k = 0; k < POS; k += 32) {
        v16bf av = load_afrag_f32(Vp, POS, mt * 16, k, lane);
#pragma unroll
        for (int j = 0; j < 8; ++j) {
            v16bf bv = load_bfrag(aw, POS, (ntBase + j) * 16, k, lane);
            acc[j] = wmma_bf16(av, bv, acc[j]);
        }
    }
#pragma unroll
    for (int j = 0; j < 8; ++j) {
        int d = (ntBase + j) * 16 + (lane & 15);
        u16* dst = attB + ((size_t)b * POS + d) * CC + h * HD;
#pragma unroll
        for (int r = 0; r < 8; ++r) {
            int cch = mt * 16 + r + (half << 3);
            dst[cch] = f2bf(acc[j][r]);
        }
    }
}

// K8: upsample pooled o-projection, add o_b and residual X (HBM-bound pass).
__global__ void final_kernel(const float* __restrict__ X,
                             const float* __restrict__ oout,
                             const float* __restrict__ o_b,
                             float* __restrict__ out) {
    size_t idx = (size_t)blockIdx.x * blockDim.x + threadIdx.x;
    if (idx >= (size_t)BB * CC * HW) return;
    int w  = (int)(idx & 63);
    int hh = (int)((idx >> 6) & 63);
    int o  = (int)((idx >> 12) & 511);
    int b  = (int)(idx >> 21);
    int p = ((hh >> 2) << 4) + (w >> 2);
    out[idx] = oout[((size_t)b * CC + o) * POS + p] + o_b[o] + X[idx];
}

// ---------------------------------------------------------------------------
extern "C" void kernel_launch(void* const* d_in, const int* in_sizes, int n_in,
                              void* d_out, int out_size, void* d_ws, size_t ws_size,
                              hipStream_t stream) {
    const float* X     = (const float*)d_in[0];
    const float* gn_w  = (const float*)d_in[1];
    const float* gn_b  = (const float*)d_in[2];
    const float* kqv_w = (const float*)d_in[3];
    const float* kqv_b = (const float*)d_in[4];
    const float* o_w   = (const float*)d_in[5];
    const float* o_b   = (const float*)d_in[6];
    float* out = (float*)d_out;

    char* ws = (char*)d_ws;
    size_t off = 0;
    auto alloc = [&](size_t bytes) { char* p = ws + off; off = (off + bytes + 255) & ~(size_t)255; return p; };

    float* pooledX = (float*)alloc((size_t)BB * POS * CC * 4);   // 16 MB  [b][p][c]
    float* chsum   = (float*)alloc((size_t)BB * CC * 4);
    float* chsumsq = (float*)alloc((size_t)BB * CC * 4);
    float* sc      = (float*)alloc((size_t)BB * CC * 4);
    float* sh      = (float*)alloc((size_t)BB * CC * 4);
    u16*   xnp     = (u16*)  alloc((size_t)BB * POS * CC * 2);   // 8 MB   [b][p][c] bf16
    u16*   wkbf    = (u16*)  alloc((size_t)KQV * CC * 2);        // 1.5 MB
    u16*   owbf    = (u16*)  alloc((size_t)CC * CC * 2);         // 0.5 MB
    float* kqvp    = (float*)alloc((size_t)BB * KQV * POS * 4);  // 50 MB  [b][ch][p]
    u16*   attnW   = (u16*)  alloc((size_t)BB * HEADS * POS * POS * 2); // 33.5 MB
    u16*   attB    = (u16*)  alloc((size_t)BB * POS * CC * 2);   // 8 MB   [b][p][ch] bf16
    float* oout    = (float*)alloc((size_t)BB * CC * POS * 4);   // 16 MB  [b][o][p]

    // weights -> bf16
    {
        size_t n = (size_t)KQV * CC + (size_t)CC * CC;
        cvt_w_kernel<<<(unsigned)((n + 255) / 256), 256, 0, stream>>>(kqv_w, o_w, wkbf, owbf);
    }
    // pooled X + per-channel stats
    pool_stats_kernel<<<BB * CC, 256, 0, stream>>>(X, pooledX, chsum, chsumsq);
    // group stats -> per-channel scale/shift
    group_stats_kernel<<<(BB * 128 + 255) / 256, 256, 0, stream>>>(chsum, chsumsq, gn_w, gn_b, sc, sh);
    // normalized pooled activations, bf16 [b][p][c]
    {
        size_t n = (size_t)BB * POS * CC;
        norm_bf16_kernel<<<(unsigned)((n + 255) / 256), 256, 0, stream>>>(pooledX, sc, sh, xnp);
    }
    // kqv GEMM at pooled res: [1536x512] x [512x256] per batch, +kqv_b
    {
        int macroTiles = (KQV / 32) * (POS / 32);   // 48*8 = 384
        gemm_bf16_kernel<<<dim3(macroTiles / 8, BB), 256, 0, stream>>>(
            wkbf, xnp, kqvp, kqv_b, KQV, POS, CC,
            (size_t)POS * CC, (size_t)KQV * POS);
    }
    // attention scores + softmax
    attn_softmax_kernel<<<dim3(2, HEADS, BB), 256, 0, stream>>>(kqvp, attnW);
    // attn x V -> attB[b][p][ch] bf16
    attn_v_kernel<<<dim3(HEADS, BB), 256, 0, stream>>>(kqvp, attnW, attB);
    // o projection at pooled res: [512x512] x [512x256] per batch
    {
        int macroTiles = (CC / 32) * (POS / 32);    // 16*8 = 128
        gemm_bf16_kernel<<<dim3(macroTiles / 8, BB), 256, 0, stream>>>(
            owbf, attB, oout, nullptr, CC, POS, CC,
            (size_t)POS * CC, (size_t)CC * POS);
    }
    // upsample + bias + residual
    {
        size_t n = (size_t)BB * CC * HW;
        final_kernel<<<(unsigned)(n / 256), 256, 0, stream>>>(X, oout, o_b, out);
    }
    (void)in_sizes; (void)n_in; (void)out_size; (void)ws_size;
}
